// MultiHeadAttention_4698694222705
// MI455X (gfx1250) — matrix operand order repaired and checked by reading, not compiled, not run on hardware
//
#include <hip/hip_runtime.h>
#include <hip/hip_bf16.h>
#include <math.h>

typedef __bf16 bf16;
typedef __attribute__((ext_vector_type(8)))  __bf16 v8bf;
typedef __attribute__((ext_vector_type(16))) __bf16 v16bf;
typedef __attribute__((ext_vector_type(8)))  float  v8f;

static constexpr int BATCH = 4;
static constexpr int SEQ   = 1024;
static constexpr int DIM   = 1024;
static constexpr int NH    = 16;
static constexpr int DKH   = 64;   // head dim

__device__ __forceinline__ v16bf combine16(v8bf lo, v8bf hi) {
  return __builtin_shufflevector(lo, hi, 0,1,2,3,4,5,6,7,8,9,10,11,12,13,14,15);
}

__device__ __forceinline__ v8f wmma_bf16(v16bf a, v16bf b, v8f c) {
  return __builtin_amdgcn_wmma_f32_16x16x32_bf16(false, a, false, b, (short)0, c, false, false);
}

// ---------------------------------------------------------------------------
// f32 -> bf16 elementwise convert (4 elems/thread)
// ---------------------------------------------------------------------------
__global__ __launch_bounds__(256) void convert_bf16(const float* __restrict__ src,
                                                    bf16* __restrict__ dst, int n) {
  int base = (blockIdx.x * blockDim.x + threadIdx.x) * 4;
  if (base + 3 < n) {
    float4 v = *(const float4*)(src + base);
    dst[base + 0] = (bf16)v.x;
    dst[base + 1] = (bf16)v.y;
    dst[base + 2] = (bf16)v.z;
    dst[base + 3] = (bf16)v.w;
  }
}

// ---------------------------------------------------------------------------
// W[k][n] (f32, DIMxDIM) -> Wt[n][k] (bf16) via LDS 32x32 tile
// ---------------------------------------------------------------------------
__global__ __launch_bounds__(256) void transpose_to_bf16(const float* __restrict__ W,
                                                         bf16* __restrict__ Wt) {
  __shared__ float tile[32][33];
  const int tx = threadIdx.x, ty = threadIdx.y;
  const int bx = blockIdx.x * 32, by = blockIdx.y * 32;
#pragma unroll
  for (int j = 0; j < 4; ++j)
    tile[ty + j * 8][tx] = W[(size_t)(by + ty + j * 8) * DIM + bx + tx];
  __syncthreads();
#pragma unroll
  for (int j = 0; j < 4; ++j)
    Wt[(size_t)(bx + ty + j * 8) * DIM + by + tx] = (bf16)tile[tx][ty + j * 8];
}

// ---------------------------------------------------------------------------
// C[4096][1024] = A[4096][1024] @ W + bias,  W given as Bt[n][k] = W^T (bf16)
// Each wave owns a 16x64 output tile (4 accumulators, A reused 4x per K-step).
// mode 0: bf16 row-major out   (Q, K projections)
// mode 1: bf16 scatter to Vt[b][h][dk][s]  (V projection, pre-transposed)
// mode 2: f32 row-major out    (final output projection)
// ---------------------------------------------------------------------------
__global__ __launch_bounds__(256) void gemm_bf16_nt(const bf16* __restrict__ A,
                                                    const bf16* __restrict__ Bt,
                                                    const float* __restrict__ bias,
                                                    void* __restrict__ out, int mode) {
  const int lane = threadIdx.x & 31;
  const int half = lane >> 4;     // 0: lanes 0-15, 1: lanes 16-31
  const int lr   = lane & 15;
  const int wave = blockIdx.x * (blockDim.x >> 5) + (threadIdx.x >> 5);
  const int m0 = (wave >> 4) * 16;      // 256 m-tiles
  const int n0 = (wave & 15) * 64;      // 16 n-groups of width 64

  v8f acc[4] = {{}, {}, {}, {}};
  const bf16* arow = A + (size_t)(m0 + lr) * DIM;

  for (int k0 = 0; k0 < DIM; k0 += 32) {
    // A tile 16x32: lanes 0-15 hold K {k0..k0+7, k0+16..k0+23}; lanes 16-31 the +8 halves
    const bf16* pa = arow + k0 + half * 8;
    v16bf a = combine16(*(const v8bf*)pa, *(const v8bf*)(pa + 16));
    __builtin_prefetch(arow + k0 + 64, 0, 3);
#pragma unroll
    for (int t = 0; t < 4; ++t) {
      // B tile 32x16: lane = column n, K {k0..k0+7, k0+16..k0+23} (+8 for upper half), matching A
      const bf16* pb = Bt + (size_t)(n0 + t * 16 + lr) * DIM + k0 + half * 8;
      v16bf b = combine16(*(const v8bf*)pb, *(const v8bf*)(pb + 16));
      acc[t] = wmma_bf16(a, b, acc[t]);
    }
  }

#pragma unroll
  for (int t = 0; t < 4; ++t) {
    const int n = n0 + t * 16 + lr;
    const float bn = bias[n];
#pragma unroll
    for (int r = 0; r < 8; ++r) {
      const int m = m0 + r + half * 8;     // D layout: VGPR r = row r (+8 for upper half)
      const float v = acc[t][r] + bn;
      if (mode == 0) {
        ((bf16*)out)[(size_t)m * DIM + n] = (bf16)v;
      } else if (mode == 1) {
        const int b = m >> 10, s = m & 1023;
        const int h = n >> 6,  dk = n & 63;
        ((bf16*)out)[((size_t)(b * NH + h) * DKH + dk) * SEQ + s] = (bf16)v;
      } else {
        ((float*)out)[(size_t)m * DIM + n] = v;
      }
    }
  }
}

// ---------------------------------------------------------------------------
// Fused attention for one (b, h, 16-row q-tile):
//   pass 1: scores(16x1024) = Q Kt * 0.125 into LDS  (8 waves x 8 key-tiles)
//   softmax: per-row max/sum via shfl_xor; write f32 attn to d_out, bf16 to LDS
//   pass 2: X(16x64) = P V  (4 waves, A from LDS, B from Vt[b][h][dk][s])
// ---------------------------------------------------------------------------
__global__ __launch_bounds__(256) void attn_fused(const bf16* __restrict__ Q,
                                                  const bf16* __restrict__ Km,
                                                  const bf16* __restrict__ Vt,
                                                  float* __restrict__ attn,
                                                  float* __restrict__ X) {
  __shared__ float s_scores[16][1028];   // padded vs 64-bank LDS
  __shared__ bf16  s_probs[16][1032];

  const int lane = threadIdx.x & 31;
  const int half = lane >> 4;
  const int lr   = lane & 15;
  const int wave = threadIdx.x >> 5;

  const int qt = blockIdx.x & 63;
  const int h  = (blockIdx.x >> 6) & 15;
  const int b  = blockIdx.x >> 10;
  const int q0 = qt * 16;

  // Q A-tiles for DK=64: two 16x32 tiles, loaded once per wave
  const bf16* qrow = Q + (size_t)(b * SEQ + q0 + lr) * DIM + h * DKH;
  v16bf aq0 = combine16(*(const v8bf*)(qrow + half * 8),
                        *(const v8bf*)(qrow + half * 8 + 16));
  v16bf aq1 = combine16(*(const v8bf*)(qrow + 32 + half * 8),
                        *(const v8bf*)(qrow + 32 + half * 8 + 16));

  for (int kt = wave; kt < 64; kt += 8) {
    const bf16* krow = Km + (size_t)(b * SEQ + kt * 16 + lr) * DIM + h * DKH + half * 8;
    v16bf b0 = combine16(*(const v8bf*)krow,        *(const v8bf*)(krow + 16));
    v16bf b1 = combine16(*(const v8bf*)(krow + 32), *(const v8bf*)(krow + 48));
    v8f acc = {};
    acc = wmma_bf16(aq0, b0, acc);
    acc = wmma_bf16(aq1, b1, acc);
#pragma unroll
    for (int r = 0; r < 8; ++r)
      s_scores[r + half * 8][kt * 16 + lr] = acc[r] * 0.125f;  // 1/sqrt(64)
  }
  __syncthreads();

  // softmax: wave w handles rows w and w+8; lane covers 32 strided columns
  for (int rr = 0; rr < 2; ++rr) {
    const int row = wave + rr * 8;
    float mx = -3.0e38f;
    for (int c = lane; c < SEQ; c += 32) mx = fmaxf(mx, s_scores[row][c]);
#pragma unroll
    for (int off = 16; off > 0; off >>= 1) mx = fmaxf(mx, __shfl_xor(mx, off, 32));
    float sum = 0.f;
    for (int c = lane; c < SEQ; c += 32) {
      float e = __expf(s_scores[row][c] - mx);
      s_scores[row][c] = e;
      sum += e;
    }
#pragma unroll
    for (int off = 16; off > 0; off >>= 1) sum += __shfl_xor(sum, off, 32);
    const float rs = 1.0f / sum;
    float* gout = attn + ((size_t)(b * NH + h) * SEQ + q0 + row) * SEQ;
    for (int c = lane; c < SEQ; c += 32) {
      float p = s_scores[row][c] * rs;
      gout[c] = p;
      s_probs[row][c] = (bf16)p;
    }
  }
  __syncthreads();

  // pass 2: P @ V, waves 0-3 each own a 16-wide d-slice
  if (wave < 4) {
    const int n0 = wave * 16;
    v8f acc = {};
    const bf16* vrow = Vt + ((size_t)(b * NH + h) * DKH + n0 + lr) * SEQ;
    for (int k0 = 0; k0 < SEQ; k0 += 32) {
      const bf16* pa = &s_probs[lr][k0 + half * 8];
      v16bf a = combine16(*(const v8bf*)pa, *(const v8bf*)(pa + 16));
      const bf16* pb = vrow + k0 + half * 8;
      v16bf bb = combine16(*(const v8bf*)pb, *(const v8bf*)(pb + 16));
      acc = wmma_bf16(a, bb, acc);
    }
#pragma unroll
    for (int r = 0; r < 8; ++r)
      X[(size_t)(b * SEQ + q0 + r + half * 8) * DIM + h * DKH + n0 + lr] = acc[r];
  }
}

// ---------------------------------------------------------------------------
// x = X + residual; LayerNorm(eps=1e-6) * gamma + beta -> bf16 (GEMM A input)
// one block (256 threads, 4 elems each) per row
// ---------------------------------------------------------------------------
__global__ __launch_bounds__(256) void resid_layernorm(const float* __restrict__ X,
                                                       const float* __restrict__ resid,
                                                       const float* __restrict__ gamma,
                                                       const float* __restrict__ beta,
                                                       bf16* __restrict__ out) {
  __shared__ float red[256];
  const int row = blockIdx.x;
  const int tid = threadIdx.x;
  float x[4];
  float sum = 0.f;
#pragma unroll
  for (int i = 0; i < 4; ++i) {
    const int c = tid + i * 256;
    x[i] = X[(size_t)row * DIM + c] + resid[(size_t)row * DIM + c];
    sum += x[i];
  }
  red[tid] = sum; __syncthreads();
  for (int s = 128; s > 0; s >>= 1) { if (tid < s) red[tid] += red[tid + s]; __syncthreads(); }
  const float mu = red[0] * (1.0f / DIM);
  __syncthreads();
  float vs = 0.f;
#pragma unroll
  for (int i = 0; i < 4; ++i) { const float d = x[i] - mu; vs += d * d; }
  red[tid] = vs; __syncthreads();
  for (int s = 128; s > 0; s >>= 1) { if (tid < s) red[tid] += red[tid + s]; __syncthreads(); }
  const float rstd = rsqrtf(red[0] * (1.0f / DIM) + 1e-6f);
#pragma unroll
  for (int i = 0; i < 4; ++i) {
    const int c = tid + i * 256;
    out[(size_t)row * DIM + c] = (bf16)((x[i] - mu) * rstd * gamma[c] + beta[c]);
  }
}

// ---------------------------------------------------------------------------
extern "C" void kernel_launch(void* const* d_in, const int* in_sizes, int n_in,
                              void* d_out, int out_size, void* d_ws, size_t ws_size,
                              hipStream_t stream) {
  (void)in_sizes; (void)n_in; (void)out_size; (void)ws_size;
  const float* query = (const float*)d_in[0];
  const float* keyi  = (const float*)d_in[1];
  const float* vali  = (const float*)d_in[2];
  const float* Wq    = (const float*)d_in[3];
  const float* bq    = (const float*)d_in[4];
  const float* Wk    = (const float*)d_in[5];
  const float* bk    = (const float*)d_in[6];
  const float* Wv    = (const float*)d_in[7];
  const float* bv    = (const float*)d_in[8];
  const float* Wo    = (const float*)d_in[9];
  const float* bo    = (const float*)d_in[10];
  const float* gamma = (const float*)d_in[11];
  const float* beta  = (const float*)d_in[12];

  const size_t NTOK = (size_t)BATCH * SEQ;          // 4096
  const size_t NEL  = NTOK * DIM;                   // 4M
  char* ws = (char*)d_ws;
  size_t off = 0;
  bf16* qbf = (bf16*)(ws + off); off += NEL * 2;    // 8 MB each
  bf16* kbf = (bf16*)(ws + off); off += NEL * 2;
  bf16* vbf = (bf16*)(ws + off); off += NEL * 2;
  bf16* WqT = (bf16*)(ws + off); off += (size_t)DIM * DIM * 2;   // 2 MB each
  bf16* WkT = (bf16*)(ws + off); off += (size_t)DIM * DIM * 2;
  bf16* WvT = (bf16*)(ws + off); off += (size_t)DIM * DIM * 2;
  bf16* WoT = (bf16*)(ws + off); off += (size_t)DIM * DIM * 2;
  bf16* Qbf = (bf16*)(ws + off); off += NEL * 2;
  bf16* Kbf = (bf16*)(ws + off); off += NEL * 2;
  bf16* Vt  = (bf16*)(ws + off); off += NEL * 2;
  float* Xf = (float*)(ws + off); off += NEL * 4;   // 16 MB
  bf16* LNx = (bf16*)(ws + off); off += NEL * 2;    // total ~80 MB

  float* out  = (float*)d_out;
  float* attn = out + NEL;                          // [B,H,S,S] region

  // 1) precision conversion + weight transposes
  convert_bf16<<<(int)(NEL / 1024), 256, 0, stream>>>(query, qbf, (int)NEL);
  convert_bf16<<<(int)(NEL / 1024), 256, 0, stream>>>(keyi,  kbf, (int)NEL);
  convert_bf16<<<(int)(NEL / 1024), 256, 0, stream>>>(vali,  vbf, (int)NEL);
  transpose_to_bf16<<<dim3(32, 32), dim3(32, 8), 0, stream>>>(Wq, WqT);
  transpose_to_bf16<<<dim3(32, 32), dim3(32, 8), 0, stream>>>(Wk, WkT);
  transpose_to_bf16<<<dim3(32, 32), dim3(32, 8), 0, stream>>>(Wv, WvT);
  transpose_to_bf16<<<dim3(32, 32), dim3(32, 8), 0, stream>>>(Wo, WoT);

  // 2) QKV projections (WMMA bf16, f32 accumulate)
  gemm_bf16_nt<<<512, 256, 0, stream>>>(qbf, WqT, bq, Qbf, 0);
  gemm_bf16_nt<<<512, 256, 0, stream>>>(kbf, WkT, bk, Kbf, 0);
  gemm_bf16_nt<<<512, 256, 0, stream>>>(vbf, WvT, bv, Vt, 1);

  // 3) fused scores + softmax + P@V  (one block per (b,h,q-tile))
  attn_fused<<<BATCH * NH * (SEQ / 16), 256, 0, stream>>>(Qbf, Kbf, Vt, attn, Xf);

  // 4) residual + LayerNorm, then output projection
  resid_layernorm<<<(int)NTOK, 256, 0, stream>>>(Xf, query, gamma, beta, LNx);
  gemm_bf16_nt<<<512, 256, 0, stream>>>(LNx, WoT, bo, out, 2);
}
